// MultiheadCrossAttention_68934225101451
// MI455X (gfx1250) — matrix operand order repaired and checked by reading, not compiled, not run on hardware
//
#include <hip/hip_runtime.h>
#include <hip/hip_bf16.h>

// ---------------------------------------------------------------------------
// MultiheadCrossAttention forward for MI455X (gfx1250, wave32, WMMA).
// B=2, M=2048, N=2048, C=1024, H=16, HD=64, T=4096.
// Pipeline: f32->f16 converts -> QKV gemm -> KV gemm -> flash attention -> proj.
// All matmuls use v_wmma_f32_16x16x32_f16 (f16 in, fp32 accumulate).
// Softmax row-max uses v_permlane16 butterflies (no LDS); row-sum uses a
// WMMA against an all-ones B matrix; exponentials are exp2 with log2(e)
// folded into the Q pre-scale.
// ---------------------------------------------------------------------------

namespace {
constexpr int kB  = 2;
constexpr int kM  = 2048;
constexpr int kN  = 2048;
constexpr int kC  = 1024;
constexpr int kH  = 16;
constexpr int kHD = 64;
constexpr int kT  = kM + kN;   // 4096
}

typedef __attribute__((ext_vector_type(16))) _Float16 v16h;
typedef __attribute__((ext_vector_type(8)))  _Float16 v8h;
typedef __attribute__((ext_vector_type(8)))  float    v8f;

// --- WMMA fragment loaders (ISA 7.12.2 layouts, wave32) ---------------------
// A fragment (16x32 f16, row-major source): lane l holds row M=l%16.
// lanes 0-15: K = [0..7] and [16..23]; lanes 16-31: K = [8..15] and [24..31].
__device__ __forceinline__ v16h frag_a(const _Float16* base, int stride) {
  const int lane = (int)(threadIdx.x & 31u);
  const _Float16* p = base + (lane & 15) * stride + ((lane < 16) ? 0 : 8);
  v8h lo = *(const v8h*)(p);
  v8h hi = *(const v8h*)(p + 16);
  return __builtin_shufflevector(lo, hi, 0,1,2,3,4,5,6,7,8,9,10,11,12,13,14,15);
}

// B fragment (32x16 f16) loaded from B^T storage ([N][K] row-major):
// lane l holds column N=l%16; lanes 0-15: K=[0..7],[16..23]; lanes 16-31: K=[8..15],[24..31].
__device__ __forceinline__ v16h frag_b(const _Float16* baseT, int stride) {
  const int lane = (int)(threadIdx.x & 31u);
  const _Float16* p = baseT + (lane & 15) * stride + ((lane < 16) ? 0 : 8);
  v8h lo = *(const v8h*)(p);
  v8h hi = *(const v8h*)(p + 16);
  return __builtin_shufflevector(lo, hi, 0,1,2,3,4,5,6,7,8,9,10,11,12,13,14,15);
}

__device__ __forceinline__ v8f wmma16(v16h a, v16h b, v8f c) {
  return __builtin_amdgcn_wmma_f32_16x16x32_f16(false, a, false, b, (short)0, c,
                                                false, false);
}

// --- xor-butterfly max over the 16-lane half-groups (rows of the D layout) --
#if __has_builtin(__builtin_amdgcn_permlane16)
__device__ __forceinline__ float xshfl16(float v, unsigned lo, unsigned hi) {
  union { float f; int i; } u; u.f = v;
  u.i = __builtin_amdgcn_permlane16(u.i, u.i, (int)lo, (int)hi, false, false);
  return u.f;
}
__device__ __forceinline__ float rowmax16(float v) {
  v = fmaxf(v, xshfl16(v, 0x67452301u, 0xEFCDAB89u));  // xor 1
  v = fmaxf(v, xshfl16(v, 0x54761032u, 0xDCFE98BAu));  // xor 2
  v = fmaxf(v, xshfl16(v, 0x32107654u, 0xBA98FEDCu));  // xor 4
  v = fmaxf(v, xshfl16(v, 0xFEDCBA98u, 0x76543210u));  // xor 8
  return v;
}
#else
__device__ __forceinline__ float rowmax16(float v) {
  for (int d = 1; d < 16; d <<= 1) v = fmaxf(v, __shfl_xor(v, d, 32));
  return v;
}
#endif

// --- elementwise f32 -> f16 -------------------------------------------------
__global__ void __launch_bounds__(256)
cvt_f32_to_f16(const float* __restrict__ in, _Float16* __restrict__ out, int n) {
  int i = (int)(blockIdx.x * blockDim.x + threadIdx.x);
  if (i < n) out[i] = (_Float16)in[i];
}

// --- transpose + convert: in [R][Ncol] f32 -> out [Ncol][R] f16 -------------
__global__ void __launch_bounds__(256)
cvt_transpose_f16(const float* __restrict__ in, _Float16* __restrict__ out,
                  int R, int Ncol) {
  int i = (int)(blockIdx.x * blockDim.x + threadIdx.x);
  if (i < R * Ncol) {
    int r = i / Ncol;
    int c = i - r * Ncol;
    out[(size_t)c * R + r] = (_Float16)in[i];
  }
}

// --- generic WMMA GEMM: C = A[4096 x KK] * B, B stored transposed [NN][KK] --
// 128x128 block tile, 8 waves of 64x32, K-step 32 staged via LDS.
// MODE 0: QKV epilogue (Q scaled by log2(e)/sqrt(HD), K -> [B][T][C],
//         V -> Vt [B][H][HD][T])
// MODE 1: KV epilogue from `data` (rows land at T-offset kM)
// MODE 2: projection epilogue, fp32 output + bias
template <int MODE>
__global__ void __launch_bounds__(256)
gemm_f16_wmma(const _Float16* __restrict__ A,
              const _Float16* __restrict__ Bt,
              const float*    __restrict__ bias,
              _Float16* __restrict__ qOut,
              _Float16* __restrict__ kOut,
              _Float16* __restrict__ vtOut,
              float*    __restrict__ fOut,
              int KK)
{
  __shared__ _Float16 Ash[128 * 40];   // 40-half padded stride (80B = 20 banks)
  __shared__ _Float16 Bsh[128 * 40];
  const int tid  = (int)threadIdx.x;
  const int wave = tid >> 5;
  const int lane = tid & 31;
  const int wm   = wave >> 2;          // 0..1 (M direction)
  const int wn   = wave & 3;           // 0..3 (N direction)
  const int m0   = (int)blockIdx.y * 128;
  const int n0   = (int)blockIdx.x * 128;

  v8f zero = {};
  v8f acc[4][2];
#pragma unroll
  for (int i = 0; i < 4; ++i)
#pragma unroll
    for (int j = 0; j < 2; ++j) acc[i][j] = zero;

#pragma unroll 1
  for (int k0 = 0; k0 < KK; k0 += 32) {
    // cooperative 128x32 stage of A and B^T (each thread: 2x two 16B chunks)
#pragma unroll
    for (int it = 0; it < 2; ++it) {
      int c   = tid + it * 256;        // 0..511
      int row = c >> 2;
      int cc  = (c & 3) * 8;
      *(v8h*)(&Ash[row * 40 + cc]) =
          *(const v8h*)(&A[(size_t)(m0 + row) * KK + k0 + cc]);
      *(v8h*)(&Bsh[row * 40 + cc]) =
          *(const v8h*)(&Bt[(size_t)(n0 + row) * KK + k0 + cc]);
    }
    __syncthreads();

    v16h af[4], bf[2];
#pragma unroll
    for (int i = 0; i < 4; ++i) af[i] = frag_a(&Ash[(wm * 64 + i * 16) * 40], 40);
#pragma unroll
    for (int j = 0; j < 2; ++j) bf[j] = frag_b(&Bsh[(wn * 32 + j * 16) * 40], 40);
#pragma unroll
    for (int i = 0; i < 4; ++i)
#pragma unroll
      for (int j = 0; j < 2; ++j)
        acc[i][j] = wmma16(af[i], bf[j], acc[i][j]);
    __syncthreads();
  }

  // epilogue: D layout = lane holds (m = r + 8*(lane>=16), n = lane%16)
  const int hl = lane >> 4;
  const int nn = lane & 15;
  const float kQScale = 0.125f * 1.44269504088896341f;   // log2(e)/sqrt(HD)
#pragma unroll
  for (int i = 0; i < 4; ++i)
#pragma unroll
    for (int j = 0; j < 2; ++j)
#pragma unroll
      for (int r = 0; r < 8; ++r) {
        const int gm = m0 + wm * 64 + i * 16 + r + hl * 8;   // 0..4095
        const int gn = n0 + wn * 32 + j * 16 + nn;
        float val = acc[i][j][r] + bias[gn];
        if (MODE == 0) {                       // qkv from x
          const int b  = gm >> 11;
          const int ml = gm & 2047;
          if (gn < kC) {
            qOut[(size_t)gm * kC + gn] = (_Float16)(val * kQScale);
          } else if (gn < 2 * kC) {
            kOut[((size_t)(b * kT + ml)) * kC + (gn - kC)] = (_Float16)val;
          } else {
            const int c = gn - 2 * kC, h = c >> 6, d = c & 63;
            vtOut[((size_t)((b * kH + h) * kHD + d)) * kT + ml] = (_Float16)val;
          }
        } else if (MODE == 1) {                // kv from data
          const int b  = gm >> 11;
          const int nl = gm & 2047;
          if (gn < kC) {
            kOut[((size_t)(b * kT + kM + nl)) * kC + gn] = (_Float16)val;
          } else {
            const int c = gn - kC, h = c >> 6, d = c & 63;
            vtOut[((size_t)((b * kH + h) * kHD + d)) * kT + (kM + nl)] =
                (_Float16)val;
          }
        } else {                               // projection, f32 out
          fOut[(size_t)gm * kC + gn] = val;
        }
      }
}

// --- flash attention: grid (M/64, B*H), 4 waves; wave owns 16 query rows ----
// Scores are already in log2 units (Q pre-scaled by log2(e)/sqrt(HD)), so all
// exponentials are raw exp2f -> v_exp_f32.
__global__ void __launch_bounds__(128)
flash_attn_wmma(const _Float16* __restrict__ Q,   // [B*M][C], pre-scaled
                const _Float16* __restrict__ K,   // [B][T][C]
                const _Float16* __restrict__ Vt,  // [B][H][HD][T]
                _Float16* __restrict__ Oh)        // [B*M][C]
{
  __shared__ _Float16 Psh[4 * 16 * 40];           // private 16x32 slab per wave
  const int tid  = (int)threadIdx.x;
  const int wave = tid >> 5;
  const int lane = tid & 31;
  const int hl   = lane >> 4;
  const int nn   = lane & 15;
  const int b    = (int)blockIdx.y >> 4;
  const int h    = (int)blockIdx.y & 15;
  const int m0   = (int)blockIdx.x * 64 + wave * 16;

  const _Float16* qbase = Q + ((size_t)(b * kM + m0)) * kC + h * kHD;
  const v16h qa0 = frag_a(qbase,      kC);        // d = 0..31
  const v16h qa1 = frag_a(qbase + 32, kC);        // d = 32..63

  v8f zero = {};
  v8f o[4];
#pragma unroll
  for (int j = 0; j < 4; ++j) o[j] = zero;
  float rmax[8], rsum[8];
#pragma unroll
  for (int r = 0; r < 8; ++r) { rmax[r] = -1.0e30f; rsum[r] = 0.0f; }

  v16h onesb;                                     // 32x16 all-ones B matrix
#pragma unroll
  for (int i = 0; i < 16; ++i) onesb[i] = (_Float16)1.0f;

  _Float16* pbase = &Psh[wave * 16 * 40];

#pragma unroll 1
  for (int t0 = 0; t0 < kT; t0 += 32) {
    // S = Q * K^T for this 16x32 tile (B-frag rows come straight from K rows)
    const _Float16* kbase = K + ((size_t)(b * kT + t0)) * kC + h * kHD;
    v8f s[2];
    s[0] = zero; s[1] = zero;
#pragma unroll
    for (int g = 0; g < 2; ++g) {
      v16h kb0 = frag_b(kbase + (size_t)(g * 16) * kC,      kC);
      v16h kb1 = frag_b(kbase + (size_t)(g * 16) * kC + 32, kC);
      s[g] = wmma16(qa0, kb0, s[g]);
      s[g] = wmma16(qa1, kb1, s[g]);
    }

    // online softmax max update (log2 domain), permlane16 butterflies
    float scale[8];
#pragma unroll
    for (int r = 0; r < 8; ++r) {
      float mloc = rowmax16(fmaxf(s[0][r], s[1][r]));
      float mnew = fmaxf(rmax[r], mloc);
      scale[r] = exp2f(rmax[r] - mnew);
      rmax[r]  = mnew;
    }

    // P = exp2(S - max), straight into the wave-private LDS transpose slab
#pragma unroll
    for (int g = 0; g < 2; ++g)
#pragma unroll
      for (int r = 0; r < 8; ++r)
        pbase[(r + hl * 8) * 40 + g * 16 + nn] =
            (_Float16)exp2f(s[g][r] - rmax[r]);
    v16h pa = frag_a(pbase, 40);                  // t-K = 0..31 (A layout)

    // row-sum of P via WMMA against ones: every column of the D tile holds
    // the row sum, which is exactly the per-lane rsum[r] state we keep.
    v8f ps = wmma16(pa, onesb, zero);
#pragma unroll
    for (int r = 0; r < 8; ++r) rsum[r] = rsum[r] * scale[r] + ps[r];
#pragma unroll
    for (int j = 0; j < 4; ++j)
#pragma unroll
      for (int r = 0; r < 8; ++r) o[j][r] = o[j][r] * scale[r];

    // O += P * V  (Vt rows are contiguous in t -> clean B-frags)
    const _Float16* vtb = Vt + ((size_t)((b * kH + h) * kHD)) * kT + t0;
#pragma unroll
    for (int j = 0; j < 4; ++j) {
      v16h vb = frag_b(vtb + (size_t)(j * 16) * kT, kT);
      o[j] = wmma16(pa, vb, o[j]);
    }
  }

  // normalize + store f16 rows (feeds projection GEMM as A)
#pragma unroll
  for (int j = 0; j < 4; ++j)
#pragma unroll
    for (int r = 0; r < 8; ++r) {
      const size_t row = (size_t)(b * kM + m0 + r + hl * 8);
      Oh[row * kC + h * kHD + j * 16 + nn] = (_Float16)(o[j][r] / rsum[r]);
    }
}

// ---------------------------------------------------------------------------
extern "C" void kernel_launch(void* const* d_in, const int* in_sizes, int n_in,
                              void* d_out, int out_size, void* d_ws, size_t ws_size,
                              hipStream_t stream) {
  (void)in_sizes; (void)n_in; (void)out_size; (void)ws_size;
  const float* x      = (const float*)d_in[0];
  const float* data   = (const float*)d_in[1];
  const float* W_attn = (const float*)d_in[2];
  const float* b_attn = (const float*)d_in[3];
  const float* W_data = (const float*)d_in[4];
  const float* b_data = (const float*)d_in[5];
  const float* W_proj = (const float*)d_in[6];
  const float* b_proj = (const float*)d_in[7];
  float* out = (float*)d_out;

  // workspace carve-up (f16 buffers), total ~76 MB
  char* ws = (char*)d_ws;
  size_t off = 0;
  _Float16* Xh  = (_Float16*)(ws + off); off += (size_t)kB * kM * kC * 2;        // 8 MB
  _Float16* Dh  = (_Float16*)(ws + off); off += (size_t)kB * kN * kC * 2;        // 8 MB
  _Float16* WAh = (_Float16*)(ws + off); off += (size_t)3 * kC * kC * 2;         // 6 MB
  _Float16* WDh = (_Float16*)(ws + off); off += (size_t)2 * kC * kC * 2;         // 4 MB
  _Float16* WPh = (_Float16*)(ws + off); off += (size_t)kC * kC * 2;             // 2 MB
  _Float16* Qb  = (_Float16*)(ws + off); off += (size_t)kB * kM * kC * 2;        // 8 MB
  _Float16* Kb  = (_Float16*)(ws + off); off += (size_t)kB * kT * kC * 2;        // 16 MB
  _Float16* Vt  = (_Float16*)(ws + off); off += (size_t)kB * kH * kHD * kT * 2;  // 16 MB
  _Float16* Oh  = (_Float16*)(ws + off); off += (size_t)kB * kM * kC * 2;        // 8 MB

  const int nXD = kB * kM * kC;  // 4,194,304
  cvt_f32_to_f16<<<(nXD + 255) / 256, 256, 0, stream>>>(x,    Xh, nXD);
  cvt_f32_to_f16<<<(nXD + 255) / 256, 256, 0, stream>>>(data, Dh, nXD);
  cvt_transpose_f16<<<(kC * 3 * kC + 255) / 256, 256, 0, stream>>>(W_attn, WAh, kC, 3 * kC);
  cvt_transpose_f16<<<(kC * 2 * kC + 255) / 256, 256, 0, stream>>>(W_data, WDh, kC, 2 * kC);
  cvt_transpose_f16<<<(kC * kC     + 255) / 256, 256, 0, stream>>>(W_proj, WPh, kC, kC);

  // QKV: [4096 x 3072] = Xh [4096 x 1024] * W_attn
  gemm_f16_wmma<0><<<dim3(3 * kC / 128, kB * kM / 128), 256, 0, stream>>>(
      Xh, WAh, b_attn, Qb, Kb, Vt, nullptr, kC);
  // KV:  [4096 x 2048] = Dh * W_data
  gemm_f16_wmma<1><<<dim3(2 * kC / 128, kB * kN / 128), 256, 0, stream>>>(
      Dh, WDh, b_data, nullptr, Kb, Vt, nullptr, kC);
  // flash attention over T=4096
  flash_attn_wmma<<<dim3(kM / 64, kB * kH), 128, 0, stream>>>(Qb, Kb, Vt, Oh);
  // projection: out = Oh * W_proj + b_proj (fp32)
  gemm_f16_wmma<2><<<dim3(kC / 128, kB * kM / 128), 256, 0, stream>>>(
      Oh, WPh, b_proj, nullptr, nullptr, nullptr, out, kC);
}
